// TFAttention_55336358642126
// MI455X (gfx1250) — matrix operand order repaired and checked by reading, not compiled, not run on hardware
//
#include <hip/hip_runtime.h>
#include <hip/hip_bf16.h>

// ---------------------------------------------------------------------------
// GPT-2 style attention block on gfx1250 (MI455X), bf16 WMMA + flash softmax.
// B=4, S=2048, D=1024, H=16, hd=64.
// ---------------------------------------------------------------------------

typedef __bf16 bf16_t;
typedef __attribute__((ext_vector_type(16))) __bf16 v16bf;
typedef __attribute__((ext_vector_type(8)))  __bf16 v8bf;
typedef __attribute__((ext_vector_type(8)))  float   v8f;
typedef __attribute__((ext_vector_type(4)))  float   v4f;

#define BATCH 4
#define SEQ   2048
#define DMODEL 1024
#define NHEAD 16
#define HDIM  64
#define TRIPLE (3 * DMODEL)

// LDS row strides (padded to dodge bank conflicts, keep 16B vector alignment)
#define SCS 36   // score rows: 36 f32 (multiple of 4 -> v4f aligned)
#define PPS 40   // prob rows: 40 bf16 (multiple of 8 -> v8bf aligned)

static __device__ __forceinline__ v8f wmma_bf16f32(v16bf a, v16bf b, v8f c) {
  // v_wmma_f32_16x16x32_bf16: D(16x16 f32) = A(16x32 bf16) * B(32x16 bf16) + C
  return __builtin_amdgcn_wmma_f32_16x16x32_bf16(false, a, false, b, (short)0, c,
                                                 false, false);
}

// A-fragment (16x32 bf16): row m = lane&15.
// elements 0..15 -> K = kchunk + half*16 + i
static __device__ __forceinline__ v16bf load_a_frag(const bf16_t* __restrict__ row,
                                                    int kchunk, int half) {
  v8bf lo = *(const v8bf*)(row + kchunk + half * 16);
  v8bf hi = *(const v8bf*)(row + kchunk + half * 16 + 8);
  return __builtin_shufflevector(lo, hi, 0,1,2,3,4,5,6,7,8,9,10,11,12,13,14,15);
}

// ---------------------------------------------------------------------------
// 1) elementwise f32 -> bf16 (coalesced both sides)
// ---------------------------------------------------------------------------
__global__ void k_convert_bf16(const float* __restrict__ in,
                               bf16_t* __restrict__ out, int n) {
  int i = blockIdx.x * blockDim.x + threadIdx.x;
  if (i < n) out[i] = (__bf16)in[i];
}

// ---------------------------------------------------------------------------
// 2) LDS-tiled transpose+convert: W[k*N+n] (f32) -> Wt[n*K+k] (bf16).
//    64x64 tiles, both global sides coalesced. N, K multiples of 64.
// ---------------------------------------------------------------------------
__global__ void k_transpose_bf16(const float* __restrict__ W,
                                 bf16_t* __restrict__ Wt, int N, int K) {
  __shared__ __bf16 tile[64 * 65];
  const int ntiles = N >> 6;
  const int nb = (blockIdx.x % ntiles) << 6;
  const int kb = (blockIdx.x / ntiles) << 6;
  const int t = threadIdx.x;            // 256 threads
#pragma unroll
  for (int i = 0; i < 16; ++i) {
    int idx = i * 256 + t;
    int kl = idx >> 6, nl = idx & 63;   // read coalesced along n
    tile[nl * 65 + kl] = (__bf16)W[(size_t)(kb + kl) * N + nb + nl];
  }
  __syncthreads();
#pragma unroll
  for (int i = 0; i < 16; ++i) {
    int idx = i * 256 + t;
    int nl = idx >> 6, kl = idx & 63;   // write coalesced along k
    Wt[(size_t)(nb + nl) * K + kb + kl] = tile[nl * 65 + kl];
  }
}

// ---------------------------------------------------------------------------
// 3) GEMM: C[M,N] = A[M,K] (bf16 row-major) * Bt[N,K]^T (bf16 N-major) + bias
//    One wave computes a 16x64 tile. OUTF32 selects f32 vs bf16 output.
// ---------------------------------------------------------------------------
template <int OUTF32>
__global__ void k_gemm_wmma(const bf16_t* __restrict__ A,
                            const bf16_t* __restrict__ Bt,
                            const float* __restrict__ bias,
                            float* __restrict__ Cf, bf16_t* __restrict__ Cb,
                            int M, int N, int K) {
  const int lane = threadIdx.x & 31;
  const int wave = threadIdx.x >> 5;
  const int w    = blockIdx.x * (blockDim.x >> 5) + wave;
  const int wavesN = N >> 6;
  const int mt = (w / wavesN) << 4;
  const int nt = (w % wavesN) << 6;
  const int half = lane >> 4;
  const int col  = lane & 15;

  v8f acc[4] = {v8f{}, v8f{}, v8f{}, v8f{}};

  const bf16_t* arow = A + (size_t)(mt + col) * K;        // A row m = lane&15
  const bf16_t* bcol = Bt + (size_t)(nt + col) * K + half * 16;

  for (int kb = 0; kb < K; kb += 32) {
    v16bf af = load_a_frag(arow, kb, half);
    v16bf b0 = *(const v16bf*)(bcol + kb);
    v16bf b1 = *(const v16bf*)(bcol + (size_t)16 * K + kb);
    v16bf b2 = *(const v16bf*)(bcol + (size_t)32 * K + kb);
    v16bf b3 = *(const v16bf*)(bcol + (size_t)48 * K + kb);
    acc[0] = wmma_bf16f32(af, b0, acc[0]);
    acc[1] = wmma_bf16f32(af, b1, acc[1]);
    acc[2] = wmma_bf16f32(af, b2, acc[2]);
    acc[3] = wmma_bf16f32(af, b3, acc[3]);
  }

#pragma unroll
  for (int sN = 0; sN < 4; ++sN) {
    int nc = nt + sN * 16 + col;
    float bv = bias[nc];
#pragma unroll
    for (int r = 0; r < 8; ++r) {
      int row = mt + r + 8 * half;     // C/D layout: VGPR r, half selects +8
      float v = acc[sN][r] + bv;
      if (OUTF32) Cf[(size_t)row * N + nc] = v;
      else        Cb[(size_t)row * N + nc] = (__bf16)v;
    }
  }
}

// ---------------------------------------------------------------------------
// 4) LDS-tiled repack of V slice of qkv into Vt[b,h,d,s]
//    (so P*V B-fragments are contiguous in s). Coalesced both sides.
// ---------------------------------------------------------------------------
__global__ void k_repack_v(const bf16_t* __restrict__ qkv,
                           bf16_t* __restrict__ vt) {
  __shared__ __bf16 tile[64 * 65];
  int blk = blockIdx.x;
  const int st = blk & 31; blk >>= 5;   // SEQ/64 = 32 s-tiles
  const int h  = blk & (NHEAD - 1);
  const int b  = blk >> 4;
  const int sbase = st << 6;
  const int t = threadIdx.x;            // 256 threads
#pragma unroll
  for (int i = 0; i < 16; ++i) {
    int idx = i * 256 + t;
    int sl = idx >> 6, d = idx & 63;    // read coalesced along d
    tile[d * 65 + sl] =
        qkv[(size_t)(b * SEQ + sbase + sl) * TRIPLE + 2 * DMODEL + h * HDIM + d];
  }
  __syncthreads();
#pragma unroll
  for (int i = 0; i < 16; ++i) {
    int idx = i * 256 + t;
    int d = idx >> 6, sl = idx & 63;    // write coalesced along s
    vt[((size_t)(b * NHEAD + h) * HDIM + d) * SEQ + sbase + sl] = tile[d * 65 + sl];
  }
}

// ---------------------------------------------------------------------------
// 5) Flash attention: one wave per (b, h, 16-row q tile). Online softmax with
//    row-per-lane reductions (2 cross-lane shuffles per 32-col chunk).
//    V fragments are issued right after the QK WMMAs (K frags just died, so
//    peak pressure stays low) and stay in flight under the whole softmax
//    section until the PV WMMAs consume them. launch_bounds(128,1) gives the
//    allocator the full VGPR file so nothing spills to scratch.
// ---------------------------------------------------------------------------
__global__ void __launch_bounds__(128, 1)
k_flash_attn(const bf16_t* __restrict__ qkv,
             const bf16_t* __restrict__ vt,
             const float* __restrict__ amask,
             bf16_t* __restrict__ aout) {
  const int lane = threadIdx.x & 31;
  const int wave = threadIdx.x >> 5;
  const int w  = blockIdx.x * 4 + wave;
  const int qt = w & 127;          // S/16 q tiles
  const int bh = w >> 7;
  const int h  = bh & (NHEAD - 1);
  const int b  = bh >> 4;
  const int half = lane >> 4;
  const int col  = lane & 15;
  const int qbase = qt * 16;

  __shared__ __align__(16) float  ldsS[4][16 * SCS];  // raw scores (f32)
  __shared__ __align__(16) __bf16 ldsP[4][16 * PPS];  // probs (bf16, A staging)
  __shared__ __align__(16) float  ldsF[4][32];        // [0..15]=sf, [16..31]=l
  float*  Sc = ldsS[wave];
  __bf16* P  = ldsP[wave];
  float*  Fx = ldsF[wave];

  // Q fragments (row m = lane&15 of this q tile), two 32-wide chunks of hd
  const bf16_t* qrow = qkv + (size_t)(b * SEQ + qbase + col) * TRIPLE + h * HDIM;
  v16bf qf0 = load_a_frag(qrow, 0, half);
  v16bf qf1 = load_a_frag(qrow, 32, half);

  const bf16_t* kbasep = qkv + (size_t)(b * SEQ) * TRIPLE + DMODEL + h * HDIM;
  const bf16_t* vbase  = vt + ((size_t)(b * NHEAD + h) * HDIM) * SEQ;

  // per-lane softmax state for the row this lane owns (row = lane&15;
  // the two halves hold identical copies, kept in sync by the xor-16 shuffles)
  float mrun = -INFINITY, lrun = 0.0f;
  const int q_own = qbase + col;

  v8f acc[4] = {v8f{}, v8f{}, v8f{}, v8f{}};

  for (int kc = 0; kc < qbase + 16; kc += 32) {
    // ---- K fragment loads, consumed immediately by the QK WMMAs
    const bf16_t* k0p = kbasep + (size_t)(kc + col) * TRIPLE + half * 16;
    const bf16_t* k1p = k0p + (size_t)16 * TRIPLE;
    v16bf kb00 = *(const v16bf*)(k0p);
    v16bf kb01 = *(const v16bf*)(k0p + 32);
    v16bf kb10 = *(const v16bf*)(k1p);
    v16bf kb11 = *(const v16bf*)(k1p + 32);

    // ---- QK^T: two 16x16 score tiles (cols kc.. and kc+16..)
    v8f s0 = {};
    s0 = wmma_bf16f32(qf0, kb00, s0);
    s0 = wmma_bf16f32(qf1, kb01, s0);
    v8f s1 = {};
    s1 = wmma_bf16f32(qf0, kb10, s1);
    s1 = wmma_bf16f32(qf1, kb11, s1);

    // ---- V fragment loads issued here (K frags just died): they stay in
    //      flight through the whole softmax and are consumed by the PV WMMAs.
    const bf16_t* vp = vbase + (size_t)col * SEQ + kc + half * 16;
    v16bf vf0 = *(const v16bf*)(vp);
    v16bf vf1 = *(const v16bf*)(vp + (size_t)16 * SEQ);
    v16bf vf2 = *(const v16bf*)(vp + (size_t)32 * SEQ);
    v16bf vf3 = *(const v16bf*)(vp + (size_t)48 * SEQ);

    // ---- scatter raw scores C-layout -> LDS row-major
#pragma unroll
    for (int r = 0; r < 8; ++r) {
      Sc[(r + 8 * half) * SCS + col]      = s0[r];
      Sc[(r + 8 * half) * SCS + 16 + col] = s1[r];
    }
    asm volatile("s_wait_dscnt 0" ::: "memory");  // cross-lane LDS RAW

    // ---- lane owns row=col; this half owns cols kc+half*16 .. +15
    float xv[16];
    const float* srow = Sc + col * SCS + half * 16;
    const float* amp  = amask + b * SEQ + kc + half * 16;
#pragma unroll
    for (int j4 = 0; j4 < 4; ++j4) {
      v4f sv = *(const v4f*)(srow + j4 * 4);
      v4f av = *(const v4f*)(amp + j4 * 4);
#pragma unroll
      for (int e = 0; e < 4; ++e) {
        int j = j4 * 4 + e;
        int k = kc + half * 16 + j;
        xv[j] = ((k <= q_own) ? sv[e] * 0.125f : -10000.0f) + av[e];
      }
    }
    float t = xv[0];
#pragma unroll
    for (int j = 1; j < 16; ++j) t = fmaxf(t, xv[j]);
    t = fmaxf(t, __shfl_xor(t, 16));            // combine the two halves
    float mnew = fmaxf(mrun, t);
    float sf = __expf(mrun - mnew);
    float p[16];
    float ps = 0.0f;
#pragma unroll
    for (int j = 0; j < 16; ++j) { p[j] = __expf(xv[j] - mnew); ps += p[j]; }
    ps += __shfl_xor(ps, 16);                   // full row sum
    lrun = lrun * sf + ps;
    mrun = mnew;

    // ---- probs -> LDS row-major (A-fragment staging), packed stores
    v8bf pv0, pv1;
#pragma unroll
    for (int e = 0; e < 8; ++e) { pv0[e] = (__bf16)p[e]; pv1[e] = (__bf16)p[8 + e]; }
    *(v8bf*)(P + col * PPS + half * 16)     = pv0;
    *(v8bf*)(P + col * PPS + half * 16 + 8) = pv1;
    if (half == 0) Fx[col] = sf;                // per-row rescale factor
    asm volatile("s_wait_dscnt 0" ::: "memory");

    // ---- rescale accumulators with per-row sf (C-layout rows r+8*half)
    v4f sfa = *(const v4f*)(Fx + 8 * half);
    v4f sfb = *(const v4f*)(Fx + 8 * half + 4);
#pragma unroll
    for (int f = 0; f < 4; ++f) {
#pragma unroll
      for (int r = 0; r < 4; ++r) {
        acc[f][r]     *= sfa[r];
        acc[f][4 + r] *= sfb[r];
      }
    }

    // ---- P (16x32) A-fragment from LDS
    v8bf plo = *(const v8bf*)(P + col * PPS + half * 16);
    v8bf phi = *(const v8bf*)(P + col * PPS + half * 16 + 8);
    v16bf pf = __builtin_shufflevector(plo, phi,
                                       0,1,2,3,4,5,6,7,8,9,10,11,12,13,14,15);

    // ---- O += P * V, hd split into 4 column tiles of 16
    acc[0] = wmma_bf16f32(pf, vf0, acc[0]);
    acc[1] = wmma_bf16f32(pf, vf1, acc[1]);
    acc[2] = wmma_bf16f32(pf, vf2, acc[2]);
    acc[3] = wmma_bf16f32(pf, vf3, acc[3]);
  }

  // ---- epilogue: redistribute denominators to C-layout, divide, merge heads
  if (half == 0) Fx[16 + col] = lrun;
  asm volatile("s_wait_dscnt 0" ::: "memory");
  v4f la = *(const v4f*)(Fx + 16 + 8 * half);
  v4f lb = *(const v4f*)(Fx + 16 + 8 * half + 4);
  float linv[8];
#pragma unroll
  for (int r = 0; r < 4; ++r) { linv[r] = 1.0f / la[r]; linv[4 + r] = 1.0f / lb[r]; }

#pragma unroll
  for (int f = 0; f < 4; ++f) {
    int nc = h * HDIM + f * 16 + col;
#pragma unroll
    for (int r = 0; r < 8; ++r) {
      int row = qbase + r + 8 * half;
      aout[(size_t)(b * SEQ + row) * DMODEL + nc] = (__bf16)(acc[f][r] * linv[r]);
    }
  }
}

// ---------------------------------------------------------------------------
// host-side launcher
// ---------------------------------------------------------------------------
extern "C" void kernel_launch(void* const* d_in, const int* in_sizes, int n_in,
                              void* d_out, int out_size, void* d_ws, size_t ws_size,
                              hipStream_t stream) {
  (void)in_sizes; (void)n_in; (void)out_size; (void)ws_size;
  const float* x      = (const float*)d_in[0];   // [4,2048,1024]
  const float* amask  = (const float*)d_in[1];   // [4,1,1,2048]
  const float* W_attn = (const float*)d_in[2];   // [1024,3072]
  const float* b_attn = (const float*)d_in[3];   // [3072]
  const float* W_proj = (const float*)d_in[4];   // [1024,1024]
  const float* b_proj = (const float*)d_in[5];   // [1024]
  float* out = (float*)d_out;                    // [4,2048,1024]

  char* ws = (char*)d_ws;
  bf16_t* xb     = (bf16_t*)(ws);                       // 16 MB
  bf16_t* wqkvt  = (bf16_t*)(ws + 16777216);            //  6 MB  [3072,1024]
  bf16_t* wprojt = (bf16_t*)(ws + 23068672);            //  2 MB  [1024,1024]
  bf16_t* qkvb   = (bf16_t*)(ws + 25165824);            // 48 MB  [8192,3072]
  bf16_t* vtb    = (bf16_t*)(ws + 75497472);            // 16 MB  [b,h,d,s]
  bf16_t* amat   = (bf16_t*)(ws + 92274688);            // 16 MB  [8192,1024]

  const int M = BATCH * SEQ;   // 8192

  // 1) convert x to bf16
  {
    int n = M * DMODEL;
    k_convert_bf16<<<(n + 255) / 256, 256, 0, stream>>>(x, xb, n);
  }
  // 2) transpose+convert weights to N-major bf16 (tiled, coalesced)
  {
    k_transpose_bf16<<<(TRIPLE / 64) * (DMODEL / 64), 256, 0, stream>>>(
        W_attn, wqkvt, TRIPLE, DMODEL);
    k_transpose_bf16<<<(DMODEL / 64) * (DMODEL / 64), 256, 0, stream>>>(
        W_proj, wprojt, DMODEL, DMODEL);
  }
  // 3) qkv = x @ W_attn + b_attn   (bf16 out)
  {
    int waves = (M / 16) * (TRIPLE / 64);
    k_gemm_wmma<0><<<waves / 8, 256, 0, stream>>>(xb, wqkvt, b_attn,
                                                  nullptr, qkvb, M, TRIPLE, DMODEL);
  }
  // 4) repack V transposed (tiled, coalesced)
  {
    k_repack_v<<<BATCH * NHEAD * (SEQ / 64), 256, 0, stream>>>(qkvb, vtb);
  }
  // 5) flash attention -> merged heads (bf16)
  {
    int waves = BATCH * NHEAD * (SEQ / 16);   // 8192
    k_flash_attn<<<waves / 4, 128, 0, stream>>>(qkvb, vtb, amask, amat);
  }
  // 6) out = a @ W_proj + b_proj   (f32 out)
  {
    int waves = (M / 16) * (DMODEL / 64);
    k_gemm_wmma<1><<<waves / 8, 256, 0, stream>>>(amat, wprojt, b_proj,
                                                  out, nullptr, M, DMODEL, DMODEL);
  }
}
